// PoliceDiscriminator_88399016887015
// MI455X (gfx1250) — matrix operand order repaired and checked by reading, not compiled, not run on hardware
//
#include <hip/hip_runtime.h>
#include <hip/hip_bf16.h>

typedef _Float16 f16;
typedef __attribute__((ext_vector_type(16))) _Float16 v16h;
typedef __attribute__((ext_vector_type(8)))  _Float16 v8h;
typedef __attribute__((ext_vector_type(8)))  float    v8f;
typedef __attribute__((ext_vector_type(4)))  float    v4f;

#define NEG_SLOPE 0.2f
#define LDS_STRIDE 136   // halves per staged row (272B, 16B-aligned, non-power-of-2 bank pattern)

// ---------------- helpers ----------------

__device__ __forceinline__ float atomicMaxFloat(float* addr, float value) {
    // IEEE trick: positive floats order as signed ints; negative floats order
    // reversed as unsigned ints. Works with -inf initialization.
    if (value >= 0.0f)
        return __int_as_float(atomicMax((int*)addr, __float_as_int(value)));
    else
        return __uint_as_float(atomicMin((unsigned int*)addr, __float_as_uint(value)));
}

__global__ __launch_bounds__(256) void fill_neg_inf_kernel(float* __restrict__ p, int n) {
    int i = blockIdx.x * blockDim.x + threadIdx.x;
    if (i < n) p[i] = __int_as_float(0xFF800000);   // -inf
}

// ---------------- layer 1: GEMM h = x @ W1 via WMMA f16 (fp32 acc) ----------------

// Pack W1 [128 x 128] (row-major, f32) into B-fragment order (f16):
// packed[((ct*4+ks)*32 + lane)*16 + i] = W1[k][col]
//   k   = ks*32 + (lane<16 ? 0 : 8) + (i<8 ? i : i+8)
//   col = ct*16 + (lane & 15)
__global__ __launch_bounds__(256) void pack_W1_kernel(const float* __restrict__ W1,
                                                      f16* __restrict__ packed) {
    int t = blockIdx.x * blockDim.x + threadIdx.x;      // 0 .. 16383
    if (t >= 8 * 4 * 32 * 16) return;
    int i    = t & 15;
    int lane = (t >> 4) & 31;
    int ks   = (t >> 9) & 3;
    int ct   = t >> 11;
    int k    = ks * 32 + ((lane < 16) ? 0 : 8) + ((i < 8) ? i : i + 8);
    int col  = ct * 16 + (lane & 15);
    packed[t] = (f16)W1[k * 128 + col];
}

// One block = 16 output rows x 128 cols; 8 waves, wave w owns 16x16 col-tile w.
__global__ __launch_bounds__(256) void gemm_h_kernel(const float* __restrict__ x,
                                                     const f16* __restrict__ packedB,
                                                     float* __restrict__ h, int N) {
    __shared__ f16 xs[16 * LDS_STRIDE];
    const int t    = threadIdx.x;
    const int row0 = blockIdx.x * 16;

    // stage 16x128 x-tile as f16 into LDS (each thread: 8 consecutive floats of one row)
    {
        int idx = t * 8;                 // 0..2040, 8 floats stay within one row
        int r   = idx >> 7;
        int c   = idx & 127;
        int rr  = row0 + r; if (rr >= N) rr = N - 1;
        v4f a0 = *(const v4f*)(x + (size_t)rr * 128 + c);
        v4f a1 = *(const v4f*)(x + (size_t)rr * 128 + c + 4);
        f16* dstp = &xs[r * LDS_STRIDE + c];
        #pragma unroll
        for (int i = 0; i < 4; ++i) { dstp[i] = (f16)a0[i]; dstp[4 + i] = (f16)a1[i]; }
    }
    __syncthreads();

    const int wave  = t >> 5;            // col tile 0..7
    const int lane  = t & 31;
    const int m     = lane & 15;         // A row
    const int kbA   = (lane < 16) ? 0 : 8;

    v8f acc = {};
    #pragma unroll
    for (int ks = 0; ks < 4; ++ks) {
        // A fragment: halves [kb..kb+7] and [kb+16..kb+23] of row m (two b128 LDS loads)
        const f16* arow = &xs[m * LDS_STRIDE + ks * 32 + kbA];
        v8h alo = *(const v8h*)(arow);
        v8h ahi = *(const v8h*)(arow + 16);
        v16h A;
        #pragma unroll
        for (int i = 0; i < 8; ++i) { A[i] = alo[i]; A[i + 8] = ahi[i]; }
        // B fragment: contiguous 32B per lane from the pre-packed W1
        v16h B = *(const v16h*)(packedB + (size_t)(((wave * 4 + ks) * 32) + lane) * 16);
        acc = __builtin_amdgcn_wmma_f32_16x16x32_f16(false, A, false, B,
                                                     (short)0, acc, false, false);
    }

    // C layout: lane -> col n = lane&15; rows r + (lane<16?0:8), r = 0..7.
    // Uniform tile-level bound check: fast path has 8 straight coalesced stores
    // (16 lanes of a row write 64 contiguous bytes); guarded path only for the
    // final partial block (never taken when N % 16 == 0).
    const int n     = wave * 16 + (lane & 15);
    const int rbase = row0 + ((lane < 16) ? 0 : 8);
    float* hp = h + (size_t)rbase * 128 + n;
    if (row0 + 16 <= N) {
        #pragma unroll
        for (int r = 0; r < 8; ++r) hp[(size_t)r * 128] = acc[r];
    } else {
        #pragma unroll
        for (int r = 0; r < 8; ++r)
            if (rbase + r < N) hp[(size_t)r * 128] = acc[r];
    }
}

// ---------------- layer 1: attention projections a_s/a_d [N,2] ----------------

__global__ __launch_bounds__(256) void attn_proj1_kernel(const float* __restrict__ h,
                                                         const float* __restrict__ att_src,
                                                         const float* __restrict__ att_dst,
                                                         float* __restrict__ a_s,
                                                         float* __restrict__ a_d, int N) {
    int node = (blockIdx.x * blockDim.x + threadIdx.x) >> 5;
    int lane = threadIdx.x & 31;
    if (node >= N) return;
    // lanes 0-15: head 0 (c 0..63), lanes 16-31: head 1 (c 64..127)
    v4f hv = *(const v4f*)(h + (size_t)node * 128 + lane * 4);
    float s = 0.f, d = 0.f;
    #pragma unroll
    for (int i = 0; i < 4; ++i) {
        s += hv[i] * att_src[lane * 4 + i];
        d += hv[i] * att_dst[lane * 4 + i];
    }
    #pragma unroll
    for (int off = 8; off >= 1; off >>= 1) {   // reduce within half-wave (per head)
        s += __shfl_xor(s, off, 32);
        d += __shfl_xor(d, off, 32);
    }
    if ((lane & 15) == 0) {
        int hd = lane >> 4;
        a_s[node * 2 + hd] = s;
        a_d[node * 2 + hd] = d;
    }
}

// ---------------- layer 1: edge pass 1 (e + segment max) ----------------

__global__ __launch_bounds__(256) void edge_max_l1_kernel(const int* __restrict__ ei,
                                                          int E, int Et,
                                                          const float* __restrict__ a_s,
                                                          const float* __restrict__ a_d,
                                                          float* __restrict__ e1,
                                                          float* __restrict__ emax1) {
    int e = blockIdx.x * blockDim.x + threadIdx.x;
    if (e >= Et) return;
    int s, d;
    if (e < E) { s = ei[e]; d = ei[E + e]; } else { s = d = e - E; }
    #pragma unroll
    for (int hd = 0; hd < 2; ++hd) {
        float v = a_s[s * 2 + hd] + a_d[d * 2 + hd];
        v = (v > 0.f) ? v : NEG_SLOPE * v;
        e1[(size_t)e * 2 + hd] = v;
        atomicMaxFloat(&emax1[d * 2 + hd], v);
    }
}

// ---------------- layer 1: edge pass 2 (exp + denom + message accumulate) ----------------

// One wave per edge: lane covers 4 channels; lanes 0-15 head 0, lanes 16-31 head 1.
__global__ __launch_bounds__(256) void edge_msg_l1_kernel(const int* __restrict__ ei,
                                                          int E, int Et,
                                                          const float* __restrict__ e1,
                                                          const float* __restrict__ emax1,
                                                          const float* __restrict__ h,
                                                          float* __restrict__ denom1,
                                                          float* __restrict__ acc) {
    int edge = (blockIdx.x * blockDim.x + threadIdx.x) >> 5;
    int lane = threadIdx.x & 31;
    if (edge >= Et) return;
    int s, d;
    if (edge < E) { s = ei[edge]; d = ei[E + edge]; } else { s = d = edge - E; }
    int hd = lane >> 4;
    float m = emax1[d * 2 + hd];
    if (!(m > -3.0e38f)) m = 0.f;                     // where(isfinite(emax), emax, 0)
    float ex = __expf(e1[(size_t)edge * 2 + hd] - m);
    if ((lane & 15) == 0) atomicAdd(&denom1[d * 2 + hd], ex);
    v4f hv = *(const v4f*)(h + (size_t)s * 128 + lane * 4);
    float* ap = acc + (size_t)d * 128 + lane * 4;
    atomicAdd(ap + 0, hv[0] * ex);
    atomicAdd(ap + 1, hv[1] * ex);
    atomicAdd(ap + 2, hv[2] * ex);
    atomicAdd(ap + 3, hv[3] * ex);
}

// ---------------- node finalize L1 (+bias, ELU) fused with L2 projection ----------------

__global__ __launch_bounds__(256) void node_finalize_kernel(const float* __restrict__ acc,
                                                            const float* __restrict__ denom1,
                                                            const float* __restrict__ b1,
                                                            const float* __restrict__ W2,
                                                            const float* __restrict__ att_src2,
                                                            const float* __restrict__ att_dst2,
                                                            float* __restrict__ g,
                                                            float* __restrict__ as2,
                                                            float* __restrict__ ad2, int N) {
    int node = (blockIdx.x * blockDim.x + threadIdx.x) >> 5;
    int lane = threadIdx.x & 31;
    if (node >= N) return;
    int hd = lane >> 4;
    float den = fmaxf(denom1[node * 2 + hd], 1e-16f);
    v4f a = *(const v4f*)(acc + (size_t)node * 128 + lane * 4);
    float part = 0.f;
    #pragma unroll
    for (int i = 0; i < 4; ++i) {
        float o = a[i] / den + b1[lane * 4 + i];
        o = (o > 0.f) ? o : (__expf(o) - 1.f);        // ELU
        part += o * W2[lane * 4 + i];                 // h2 @ W2 (W2: [128,1])
    }
    #pragma unroll
    for (int off = 16; off >= 1; off >>= 1) part += __shfl_xor(part, off, 32);
    if (lane == 0) {
        g[node]   = part;
        as2[node] = part * att_src2[0];
        ad2[node] = part * att_dst2[0];
    }
}

// ---------------- layer 2: edge passes (scalar) ----------------

__global__ __launch_bounds__(256) void edge_max_l2_kernel(const int* __restrict__ ei,
                                                          int E, int Et,
                                                          const float* __restrict__ as2,
                                                          const float* __restrict__ ad2,
                                                          float* __restrict__ e2,
                                                          float* __restrict__ emax2) {
    int e = blockIdx.x * blockDim.x + threadIdx.x;
    if (e >= Et) return;
    int s, d;
    if (e < E) { s = ei[e]; d = ei[E + e]; } else { s = d = e - E; }
    float v = as2[s] + ad2[d];
    v = (v > 0.f) ? v : NEG_SLOPE * v;
    e2[e] = v;
    atomicMaxFloat(&emax2[d], v);
}

__global__ __launch_bounds__(256) void edge_msg_l2_kernel(const int* __restrict__ ei,
                                                          int E, int Et,
                                                          const float* __restrict__ e2,
                                                          const float* __restrict__ emax2,
                                                          const float* __restrict__ g,
                                                          float* __restrict__ denom2,
                                                          float* __restrict__ num2) {
    int e = blockIdx.x * blockDim.x + threadIdx.x;
    if (e >= Et) return;
    int s, d;
    if (e < E) { s = ei[e]; d = ei[E + e]; } else { s = d = e - E; }
    float m = emax2[d];
    if (!(m > -3.0e38f)) m = 0.f;
    float ex = __expf(e2[e] - m);
    atomicAdd(&denom2[d], ex);
    atomicAdd(&num2[d], ex * g[s]);
}

__global__ __launch_bounds__(256) void node_out_kernel(const float* __restrict__ num2,
                                                       const float* __restrict__ denom2,
                                                       const float* __restrict__ b2,
                                                       float* __restrict__ out, int N) {
    int n = blockIdx.x * blockDim.x + threadIdx.x;
    if (n >= N) return;
    float v = num2[n] / fmaxf(denom2[n], 1e-16f) + b2[0];
    out[n] = 1.f / (1.f + __expf(-v));
}

// ---------------- launch ----------------

extern "C" void kernel_launch(void* const* d_in, const int* in_sizes, int n_in,
                              void* d_out, int out_size, void* d_ws, size_t ws_size,
                              hipStream_t stream) {
    (void)n_in; (void)out_size; (void)ws_size;
    const float* x   = (const float*)d_in[0];
    const int*   ei  = (const int*)d_in[1];          // int32 (JAX x64 disabled)
    const float* W1  = (const float*)d_in[2];
    const float* as1 = (const float*)d_in[3];
    const float* ad1 = (const float*)d_in[4];
    const float* b1  = (const float*)d_in[5];
    const float* W2  = (const float*)d_in[6];
    const float* as2c= (const float*)d_in[7];
    const float* ad2c= (const float*)d_in[8];
    const float* b2  = (const float*)d_in[9];
    float* out = (float*)d_out;

    const int N  = in_sizes[0] / 128;
    const int E  = in_sizes[1] / 2;
    const int Et = E + N;

    // carve workspace (256B-aligned chunks)
    char*  ws  = (char*)d_ws;
    size_t off = 0;
    auto alloc = [&](size_t bytes) -> char* {
        char* p = ws + off;
        off = (off + bytes + 255) & ~(size_t)255;
        return p;
    };
    float* h       = (float*)alloc((size_t)N * 128 * 4);
    float* acc     = (float*)alloc((size_t)N * 128 * 4);
    float* e1      = (float*)alloc((size_t)Et * 2 * 4);
    float* a_s     = (float*)alloc((size_t)N * 2 * 4);
    float* a_d     = (float*)alloc((size_t)N * 2 * 4);
    float* emax1   = (float*)alloc((size_t)N * 2 * 4);
    float* denom1  = (float*)alloc((size_t)N * 2 * 4);
    float* g       = (float*)alloc((size_t)N * 4);
    float* as2     = (float*)alloc((size_t)N * 4);
    float* ad2     = (float*)alloc((size_t)N * 4);
    float* e2      = (float*)alloc((size_t)Et * 4);
    float* emax2   = (float*)alloc((size_t)N * 4);
    float* denom2  = (float*)alloc((size_t)N * 4);
    float* num2    = (float*)alloc((size_t)N * 4);
    f16*   packedB = (f16*)alloc((size_t)8 * 4 * 32 * 16 * 2);

    // init accumulators (graph-capture-safe)
    hipMemsetAsync(acc,    0, (size_t)N * 128 * 4, stream);
    hipMemsetAsync(denom1, 0, (size_t)N * 2 * 4,  stream);
    hipMemsetAsync(denom2, 0, (size_t)N * 4,      stream);
    hipMemsetAsync(num2,   0, (size_t)N * 4,      stream);
    fill_neg_inf_kernel<<<(N * 2 + 255) / 256, 256, 0, stream>>>(emax1, N * 2);
    fill_neg_inf_kernel<<<(N + 255) / 256, 256, 0, stream>>>(emax2, N);

    // layer 1 GEMM (WMMA)
    pack_W1_kernel<<<64, 256, 0, stream>>>(W1, packedB);
    gemm_h_kernel<<<(N + 15) / 16, 256, 0, stream>>>(x, packedB, h, N);

    // layer 1 attention + message passing
    attn_proj1_kernel<<<(N * 32 + 255) / 256, 256, 0, stream>>>(h, as1, ad1, a_s, a_d, N);
    edge_max_l1_kernel<<<(Et + 255) / 256, 256, 0, stream>>>(ei, E, Et, a_s, a_d, e1, emax1);
    edge_msg_l1_kernel<<<((size_t)Et * 32 + 255) / 256, 256, 0, stream>>>(
        ei, E, Et, e1, emax1, h, denom1, acc);

    // finalize layer 1 + project for layer 2
    node_finalize_kernel<<<(N * 32 + 255) / 256, 256, 0, stream>>>(
        acc, denom1, b1, W2, as2c, ad2c, g, as2, ad2, N);

    // layer 2 (scalar messages)
    edge_max_l2_kernel<<<(Et + 255) / 256, 256, 0, stream>>>(ei, E, Et, as2, ad2, e2, emax2);
    edge_msg_l2_kernel<<<(Et + 255) / 256, 256, 0, stream>>>(ei, E, Et, e2, emax2, g, denom2, num2);
    node_out_kernel<<<(N + 255) / 256, 256, 0, stream>>>(num2, denom2, b2, out, N);
}
